// RefineLoss_15418932592815
// MI455X (gfx1250) — matrix operand order repaired and checked by reading, not compiled, not run on hardware
//
#include <hip/hip_runtime.h>
#include <hip/hip_bf16.h>

// ---------------- problem constants (from reference) ----------------
#define HW_   409600     // 640*640
#define W_    640
#define H_    640
#define NROI  160
#define NCLS_ 9
#define NLVL  12
#define GCH   8          // CP-1 usable target channels
#define ROI_  28
#define IOU_L_ 0.1f
#define IOU_H_ 0.3f
#define POSW_  2.0f
#define THETA_ 0.1f

typedef __attribute__((ext_vector_type(16))) _Float16 v16h;
typedef __attribute__((ext_vector_type(8)))  float    v8f;

// ws (float) layout:
//   [0..15]    tsize[b*8+g]
//   [16..175]  iou_max[n]
//   [176..335] wght[n]
//   [336..495] seg_loss[n]
//   [512..]    int region: cls[160], gt_ch[160], pos[160]

__device__ __forceinline__ float logsig(float x) {
    // numerically stable log(sigmoid(x)) == jax.nn.log_sigmoid
    return fminf(x, 0.f) - log1pf(expf(-fabsf(x)));
}

// ---------------- kernel A: tsize[b,g] = sum over image of targets ----------------
__global__ void k_tsize(const float* __restrict__ tgt, float* __restrict__ ws) {
    __shared__ float sm[256];
    const int blk = blockIdx.x;            // 0..15
    const int b = blk >> 3, g = blk & 7;
    const float* p = tgt + (size_t)(b * NCLS_ + 1 + g) * HW_;
    float s = 0.f;
    for (int i = threadIdx.x; i < HW_; i += 256) s += p[i];
    sm[threadIdx.x] = s;
    __syncthreads();
    for (int off = 128; off > 0; off >>= 1) {
        if (threadIdx.x < (unsigned)off) sm[threadIdx.x] += sm[threadIdx.x + off];
        __syncthreads();
    }
    if (threadIdx.x == 0) ws[blk] = sm[0];
}

// ---------------- kernel B: per-ROI interp/psum via WMMA, then iou/cls/wght ----------------
// One wave32 per ROI. D = A(16x32 f16) x B(32x16 f16) + C(16x16 f32).
//   A rows 0..7 = targets channels, row 8 = ones (psum), rows 9..15 = 0,
//   realized branch-free as A = tv*scale + bias with per-lane scale/bias.
//   B columns all equal = pred (labelI[b,lv]) over the same 32 box pixels.
// K = 32 consecutive columns of one box row. Full chunks take an unpredicated
// fast path (loads at constant immediate offsets); only the row tail is masked.
__global__ void __launch_bounds__(32)
k_interp(const float* __restrict__ labelI, const float* __restrict__ tgt,
         const int* __restrict__ bboxes, const int* __restrict__ tids,
         const float* __restrict__ clsw, float* __restrict__ ws) {
    __shared__ float sm[32 * 8];
    const int n    = blockIdx.x;
    const int lane = threadIdx.x;
    const int* bb  = bboxes + n * 7;
    const int b  = bb[0], x0 = bb[1], y0 = bb[2], x1 = bb[3], y1 = bb[4], lvl = bb[5];
    const int w = x1 - x0;
    const int h = y1 - y0;

    // base pointers at box origin
    const float* pred  = labelI + (size_t)(b * NLVL + lvl) * HW_ + (size_t)y0 * W_ + x0;
    const float* tbase = tgt    + (size_t)(b * NCLS_ + 1) * HW_  + (size_t)y0 * W_ + x0;

    const int  m  = lane & 15;
    const bool hi = lane >= 16;
    // lanes m<8 read their channel; m>=8 read channel 0 but scale it to 0.
    const float* trow0 = tbase + (size_t)((m < GCH) ? m : 0) * HW_;
    const float scale = (m < GCH) ? 1.f : 0.f;    // row 8..15 -> 0*value
    const float bias  = (m == GCH) ? 1.f : 0.f;   // row 8 -> ones (psum row)
    const int kb0 = hi ? 16 : 0;

    v8f c = {};
    for (int y = 0; y < h; ++y) {
        const float* pr = pred  + (size_t)y * W_;
        const float* tr = trow0 + (size_t)y * W_;
        int xb = 0;
        // ---------- fast path: fully in-bounds 32-wide chunks ----------
        for (; xb + 32 <= w; xb += 32) {
            const float* prc = pr + xb;
            const float* trc = tr + xb;
            v16h A, B;
#pragma unroll
            for (int e = 0; e < 16; ++e)
                B[e] = (_Float16)prc[((e < 8) ? e : e + 8) + (hi ? 8 : 0)];
#pragma unroll
            for (int j = 0; j < 8; ++j) {
                int kb = ((j < 4) ? 2 * j : 16 + 2 * (j - 4)) + (hi ? 8 : 0);
#pragma unroll
                for (int t = 0; t < 2; ++t)
                    A[2 * j + t] = (_Float16)(trc[kb + t] * scale + bias);
            }
            c = __builtin_amdgcn_wmma_f32_16x16x32_f16(false, A, false, B,
                                                       (short)0, c, false, false);
        }
        // ---------- tail: one masked chunk per row ----------
        if (xb < w) {
            v16h A, B;
#pragma unroll
            for (int e = 0; e < 16; ++e) {
                int k  = xb + ((e < 8) ? e : e + 8) + (hi ? 8 : 0);
                int kc = (k < w) ? k : 0;
                float v = pr[kc];
                B[e] = (_Float16)((k < w) ? v : 0.f);
            }
#pragma unroll
            for (int j = 0; j < 8; ++j) {
                int kb = ((j < 4) ? 2 * j : 16 + 2 * (j - 4)) + (hi ? 8 : 0);
#pragma unroll
                for (int t = 0; t < 2; ++t) {
                    int  k     = xb + kb + t;
                    bool valid = k < w;
                    int  kc    = valid ? k : 0;
                    float tv   = tr[kc];
                    float v    = valid ? (tv * scale + bias) : 0.f;
                    A[2 * j + t] = (_Float16)v;
                }
            }
            c = __builtin_amdgcn_wmma_f32_16x16x32_f16(false, A, false, B,
                                                       (short)0, c, false, false);
        }
    }

    // C/D layout: lanes 0-15 -> M=r (VGPR r), N=lane; lanes 16-31 -> M=8+r.
#pragma unroll
    for (int r = 0; r < 8; ++r) sm[lane * 8 + r] = c[r];
    __syncthreads();

    if (lane == 0) {
        float psum = sm[16 * 8 + 0];   // lane16 VGPR0 == D[M=8,N=0]
        float best = -1.f;
        int   bg   = 0;
#pragma unroll
        for (int g = 0; g < GCH; ++g) {
            float inter = sm[g];                 // lane0 VGPR g == D[M=g,N=0]
            float ts    = ws[b * GCH + g];
            float iou   = inter / (ts + psum - inter + 1.0f);
            if (iou > best) { best = iou; bg = g; }   // first-max like jnp.argmax
        }
        bool pos = best >= IOU_H_;
        bool neg = (best > 0.f) && (best < IOU_L_);
        int  cls = pos ? tids[b * NCLS_ + 1 + bg] : 0;
        float wg = pos ? (clsw[cls] + clsw[0]) : (neg ? clsw[0] : 0.f);

        int* ints = (int*)(ws + 512);
        ws[16 + n]  = best;
        ws[176 + n] = wg;
        ints[n]       = cls;
        ints[160 + n] = bg;
        ints[320 + n] = pos ? 1 : 0;
    }
}

// ---------------- kernel C: roi_align(targets[b,gt_ch]) -> mask_gt -> seg loss ----------------
__global__ void k_seg(const float* __restrict__ obj, const float* __restrict__ tgt,
                      const int* __restrict__ bboxes, float* __restrict__ ws) {
    __shared__ float sm[256];
    const int n  = blockIdx.x;
    const int* bb = bboxes + n * 7;
    const int b  = bb[0];
    const float x0 = (float)bb[1], y0 = (float)bb[2];
    const float x1 = (float)bb[3], y1 = (float)bb[4];
    const int* ints = (const int*)(ws + 512);
    const int gch = ints[160 + n];
    const int pos = ints[320 + n];
    const float* t = tgt + (size_t)(b * NCLS_ + 1 + gch) * HW_;
    const float rw = fmaxf(x1 - x0, 1.f), rh = fmaxf(y1 - y0, 1.f);

    float s = 0.f;
    for (int i = threadIdx.x; i < ROI_ * ROI_; i += 256) {
        int ox = i % ROI_, oy = i / ROI_;
        float xs = x0 + (ox + 0.5f) * (rw / ROI_);
        float ys = y0 + (oy + 0.5f) * (rh / ROI_);
        xs = fminf(fmaxf(xs, 0.f), (float)(W_ - 1));
        ys = fminf(fmaxf(ys, 0.f), (float)(H_ - 1));
        int xi = (int)floorf(xs), yi = (int)floorf(ys);
        int xi1 = xi + 1; if (xi1 > W_ - 1) xi1 = W_ - 1;
        int yi1 = yi + 1; if (yi1 > H_ - 1) yi1 = H_ - 1;
        float lx = xs - (float)xi, ly = ys - (float)yi;
        float v00 = t[yi  * W_ + xi], v01 = t[yi  * W_ + xi1];
        float v10 = t[yi1 * W_ + xi], v11 = t[yi1 * W_ + xi1];
        float v = v00 * (1.f - ly) * (1.f - lx) + v01 * (1.f - ly) * lx
                + v10 * ly * (1.f - lx)        + v11 * ly * lx;
        float mgt = (pos && v >= 0.5f) ? 1.f : 0.f;
        float x = obj[n * ROI_ * ROI_ + i];
        s += -(POSW_ * mgt * logsig(x) + (1.f - mgt) * logsig(-x));
    }
    sm[threadIdx.x] = s;
    __syncthreads();
    for (int off = 128; off > 0; off >>= 1) {
        if (threadIdx.x < (unsigned)off) sm[threadIdx.x] += sm[threadIdx.x + off];
        __syncthreads();
    }
    if (threadIdx.x == 0) ws[336 + n] = sm[0] / (float)(ROI_ * ROI_);
}

// ---------------- kernel D: per-ROI losses + weighted scalar reductions ----------------
__global__ void k_final(const float* __restrict__ logits, const float* __restrict__ iou_s,
                        const float* __restrict__ clsw, const float* __restrict__ ws,
                        float* __restrict__ out) {
    __shared__ float s0[256], s1[256], s2[256], s3[256];
    const int t = threadIdx.x;
    float sw = 0.f, si = 0.f, sc = 0.f, ss = 0.f;
    if (t < NROI) {
        float wg = ws[176 + t];
        // smooth-L1-ish iou loss
        float d  = fabsf(iou_s[t] - ws[16 + t]);
        float il = (d < THETA_) ? (d * d * (0.5f / THETA_)) : (d - THETA_ + THETA_ * 0.5f);
        // weighted NLL over log_softmax
        const float* lg = logits + t * NCLS_;
        float mx = lg[0];
#pragma unroll
        for (int j = 1; j < NCLS_; ++j) mx = fmaxf(mx, lg[j]);
        float se = 0.f;
#pragma unroll
        for (int j = 0; j < NCLS_; ++j) se += expf(lg[j] - mx);
        float lse = mx + logf(se);
        int   c   = ((const int*)(ws + 512))[t];
        float cl  = -(clsw[c] * (lg[c] - lse));
        sw = wg; si = il * wg; sc = cl * wg; ss = ws[336 + t] * wg;
    }
    s0[t] = sw; s1[t] = si; s2[t] = sc; s3[t] = ss;
    __syncthreads();
    for (int off = 128; off > 0; off >>= 1) {
        if (t < off) {
            s0[t] += s0[t + off]; s1[t] += s1[t + off];
            s2[t] += s2[t + off]; s3[t] += s3[t + off];
        }
        __syncthreads();
    }
    if (t == 0) {
        float wsum = s0[0] + 1e-4f;
        out[0] = s1[0] / wsum;
        out[1] = s2[0] / wsum;
        out[2] = s3[0] / wsum;
    }
}

// ---------------- host launcher ----------------
extern "C" void kernel_launch(void* const* d_in, const int* in_sizes, int n_in,
                              void* d_out, int out_size, void* d_ws, size_t ws_size,
                              hipStream_t stream) {
    const float* obj    = (const float*)d_in[0];   // (160,1,28,28)
    const float* logits = (const float*)d_in[1];   // (160,9)
    const float* iou_s  = (const float*)d_in[2];   // (160,1)
    const float* labelI = (const float*)d_in[3];   // (2,12,640,640)
    const float* tgt    = (const float*)d_in[4];   // (2,9,640,640)
    const float* clsw   = (const float*)d_in[5];   // (9,)
    const int*   bboxes = (const int*)d_in[6];     // (160,7)
    const int*   tids   = (const int*)d_in[7];     // (2,9)
    float* ws  = (float*)d_ws;
    float* out = (float*)d_out;

    k_tsize <<<16,   256, 0, stream>>>(tgt, ws);
    k_interp<<<NROI, 32,  0, stream>>>(labelI, tgt, bboxes, tids, clsw, ws);
    k_seg   <<<NROI, 256, 0, stream>>>(obj, tgt, bboxes, ws);
    k_final <<<1,    256, 0, stream>>>(logits, iou_s, clsw, ws, out);
}
